// DecoderMultiHeadAttention_80736795231128
// MI455X (gfx1250) — matrix operand order repaired and checked by reading, not compiled, not run on hardware
//
#include <hip/hip_runtime.h>

typedef __attribute__((ext_vector_type(16))) _Float16 v16h;
typedef __attribute__((ext_vector_type(8)))  float    v8f;

constexpr int kT = 2048, kC = 768, kH = 12, kD = 64;
constexpr int kQKV = 3 * kC;                               // 2304
constexpr size_t kHeadElems = (size_t)4 * kH * kT * kD;    // 6,291,456
constexpr size_t kXElems    = (size_t)4 * kT * kC;         // 6,291,456

union alignas(32) V16U { v16h v; _Float16 h[16]; };

__device__ inline v8f wmma16(v16h a, v16h b, v8f c) {
  return __builtin_amdgcn_wmma_f32_16x16x32_f16(false, a, false, b, (short)0, c,
                                                false, false);
}

// 16 f16 elements as two 16-byte groups of 8 (both pointers 16B-aligned)
__device__ inline v16h load2x8(const _Float16* p0, const _Float16* p1) {
  V16U u;
  *reinterpret_cast<float4*>(&u.h[0]) = *reinterpret_cast<const float4*>(p0);
  *reinterpret_cast<float4*>(&u.h[8]) = *reinterpret_cast<const float4*>(p1);
  return u.v;
}

// ---------------------------------------------------------------------------
// Prep kernels: one-time f32 -> f16 convert (x) and convert+transpose (W).
// ---------------------------------------------------------------------------
__global__ __launch_bounds__(256) void cvt_f16_kernel(
    const float* __restrict__ src, _Float16* __restrict__ dst, int n8) {
  const int gid = blockIdx.x * 256 + threadIdx.x;  // 8 elements per thread
  if (gid >= n8) return;
  const float4 f0 = reinterpret_cast<const float4*>(src)[gid * 2];
  const float4 f1 = reinterpret_cast<const float4*>(src)[gid * 2 + 1];
  alignas(16) _Float16 tmp[8] = {
      (_Float16)f0.x, (_Float16)f0.y, (_Float16)f0.z, (_Float16)f0.w,
      (_Float16)f1.x, (_Float16)f1.y, (_Float16)f1.z, (_Float16)f1.w};
  reinterpret_cast<float4*>(dst)[gid] = *reinterpret_cast<const float4*>(tmp);
}

// WT[n, k] = (f16) W[k, n]  (W is K x N row-major; WT is N x K row-major)
__global__ __launch_bounds__(256) void transpose_cvt_kernel(
    const float* __restrict__ W, _Float16* __restrict__ WT, int K, int N) {
  const int gid = blockIdx.x * 256 + threadIdx.x;
  if (gid >= K * N) return;
  const int n = gid / K, k = gid - n * K;  // consecutive threads -> consecutive k
  WT[(size_t)n * K + k] = (_Float16)W[(size_t)k * N + n];
}

// ---------------------------------------------------------------------------
// Kernel 1: qkv = x @ W_att (f16 WMMA, f32 accum), fused RoPE on q/k.
// One wave -> 32 rows x 64 cols (one full head-slice). grid=(256,9), block 128.
// q/k stored (b,h,t,d) f16 row-major; v stored transposed (b,h,d,t) f16.
// ---------------------------------------------------------------------------
__global__ __launch_bounds__(128) void qkv_rope_kernel(
    const _Float16* __restrict__ xh, const float* __restrict__ fcos,
    const float* __restrict__ fsin, const _Float16* __restrict__ WattT,
    _Float16* __restrict__ qws, _Float16* __restrict__ kws,
    _Float16* __restrict__ vtws) {
  const int lane  = threadIdx.x & 31;
  const int wave  = threadIdx.x >> 5;
  const int col16 = lane & 15;
  const int half8 = (lane >> 4) * 8;
  const int hh16  = (lane >> 4) * 16;

  const int hc    = blockIdx.y * 4 + wave;  // 0..35 head-column (64 wide)
  const int which = hc / kH;                // jnp.split order: 0=k, 1=q, 2=v
  const int h     = hc % kH;
  const int g0    = blockIdx.x * 32;        // row base in flattened (B*T)
  const int b     = g0 >> 11;               // T = 2048
  const int t0    = g0 & (kT - 1);
  const int nb    = hc * kD;

  v8f acc[2][4] = {};
  const _Float16* arow0 = xh + (size_t)(g0 + col16) * kC;
  const _Float16* arow1 = xh + (size_t)(g0 + 16 + col16) * kC;
  for (int kb = 0; kb < kC; kb += 32) {
    const v16h a0 = load2x8(arow0 + kb + half8, arow0 + kb + 16 + half8);
    const v16h a1 = load2x8(arow1 + kb + half8, arow1 + kb + 16 + half8);
#pragma unroll
    for (int nt = 0; nt < 4; ++nt) {
      const _Float16* wp =
          WattT + (size_t)(nb + nt * 16 + col16) * kC + kb + half8;
      const v16h bf = load2x8(wp, wp + 16);
      acc[0][nt] = wmma16(a0, bf, acc[0][nt]);
      acc[1][nt] = wmma16(a1, bf, acc[1][nt]);
    }
  }

  const size_t headBase = ((size_t)(b * kH + h)) * kT;
#pragma unroll
  for (int mt = 0; mt < 2; ++mt) {
    const int t0m = t0 + mt * 16;
    if (which == 2) {
      // V: store transposed (d, t) so PV B-fragments are contiguous in t.
#pragma unroll
      for (int a = 0; a < 4; ++a) {
        const int d = a * 16 + col16;
        alignas(16) _Float16 tmp[8];
#pragma unroll
        for (int i = 0; i < 8; ++i) tmp[i] = (_Float16)acc[mt][a][i];
        const size_t idx =
            ((size_t)((b * kH + h) * kD + d)) * kT + t0m + half8;
        *reinterpret_cast<float4*>(vtws + idx) =
            *reinterpret_cast<const float4*>(tmp);
      }
    } else {
      _Float16* dst = (which == 1) ? qws : kws;
      // RoPE: pair (d, d+32) lives in accumulators (a, a+2), same lane/slot.
#pragma unroll
      for (int a = 0; a < 2; ++a) {
        const int j = a * 16 + col16;  // 0..31 -> freq index
#pragma unroll
        for (int i = 0; i < 8; ++i) {
          const int t = t0m + half8 + i;
          const float c = fcos[t * 32 + j];
          const float s = fsin[t * 32 + j];
          const float e = acc[mt][a][i], o = acc[mt][a + 2][i];
          acc[mt][a][i]     = e * c - o * s;
          acc[mt][a + 2][i] = o * c + e * s;
        }
      }
#pragma unroll
      for (int a = 0; a < 4; ++a) {
        const int d = a * 16 + col16;
#pragma unroll
        for (int i = 0; i < 8; ++i) {
          const int t = t0m + half8 + i;
          dst[(headBase + t) * kD + d] = (_Float16)acc[mt][a][i];
        }
      }
    }
  }
}

// ---------------------------------------------------------------------------
// Kernel 2: causal flash attention. One wave (block of 32) handles 16 query
// rows of one (b,h). Key tiles of 32; online softmax; P converted C->A layout
// through LDS. grid = (128, 48), block 32.
// ---------------------------------------------------------------------------
__global__ __launch_bounds__(32) void attn_kernel(
    const _Float16* __restrict__ qws, const _Float16* __restrict__ kws,
    const _Float16* __restrict__ vtws, _Float16* __restrict__ attn) {
  __shared__ alignas(16) _Float16 lds[16 * 40];  // 16 rows, stride 40 f16

  const int lane  = threadIdx.x & 31;
  const int col16 = lane & 15;
  const int half8 = (lane >> 4) * 8;
  const int hh16  = (lane >> 4) * 16;
  const int qb    = blockIdx.x * 16;
  const int bh    = blockIdx.y;  // b*12 + h
  const int b     = bh / kH, h = bh % kH;

  const _Float16* qbase = qws + (size_t)bh * kT * kD;
  const _Float16* kbase = kws + (size_t)bh * kT * kD;
  const _Float16* vbase = vtws + (size_t)bh * kD * kT;

  v8f O[4] = {};
  float mrun[8], lrun[8];
#pragma unroll
  for (int i = 0; i < 8; ++i) { mrun[i] = -1e30f; lrun[i] = 0.0f; }
  const float scale = 0.125f;  // 1/sqrt(64)

  // Q fragments (reused across all key tiles): K-dim = d, split d=0..31, 32..63
  const _Float16* qrow = qbase + (size_t)(qb + col16) * kD;
  const v16h aq0 = load2x8(qrow + half8, qrow + 16 + half8);
  const v16h aq1 = load2x8(qrow + 32 + half8, qrow + 48 + half8);

  for (int ks = 0; ks < qb + 16; ks += 32) {
    // ---- scores: S[nt] covers keys ks+16*nt .. +15 --------------------
    v8f S[2] = {};
#pragma unroll
    for (int nt = 0; nt < 2; ++nt) {
      const _Float16* krow = kbase + (size_t)(ks + nt * 16 + col16) * kD;
      S[nt] = wmma16(aq0, load2x8(krow + half8, krow + 16 + half8), S[nt]);
      S[nt] = wmma16(aq1, load2x8(krow + 32 + half8, krow + 48 + half8), S[nt]);
    }
    // scale + causal mask
#pragma unroll
    for (int nt = 0; nt < 2; ++nt) {
      const int tk = ks + nt * 16 + col16;
#pragma unroll
      for (int i = 0; i < 8; ++i) {
        const int tq = qb + half8 + i;
        const float sv = S[nt][i] * scale;
        S[nt][i] = (tk <= tq) ? sv : -1e30f;
      }
    }
    // ---- online softmax ----------------------------------------------
    float mnew[8], alpha[8];
#pragma unroll
    for (int i = 0; i < 8; ++i) {
      float v = fmaxf(S[0][i], S[1][i]);
      v = fmaxf(v, __shfl_xor(v, 1, 32));
      v = fmaxf(v, __shfl_xor(v, 2, 32));
      v = fmaxf(v, __shfl_xor(v, 4, 32));
      v = fmaxf(v, __shfl_xor(v, 8, 32));
      mnew[i]  = fmaxf(mrun[i], v);
      alpha[i] = __expf(mrun[i] - mnew[i]);
    }
#pragma unroll
    for (int nt = 0; nt < 2; ++nt)
#pragma unroll
      for (int i = 0; i < 8; ++i) S[nt][i] = __expf(S[nt][i] - mnew[i]);
#pragma unroll
    for (int i = 0; i < 8; ++i) {
      float r = S[0][i] + S[1][i];
      r += __shfl_xor(r, 1, 32);
      r += __shfl_xor(r, 2, 32);
      r += __shfl_xor(r, 4, 32);
      r += __shfl_xor(r, 8, 32);
      lrun[i] = lrun[i] * alpha[i] + r;
      mrun[i] = mnew[i];
    }
#pragma unroll
    for (int a = 0; a < 4; ++a)
#pragma unroll
      for (int i = 0; i < 8; ++i) O[a][i] *= alpha[i];

    // ---- P: C layout -> A layout via LDS ------------------------------
#pragma unroll
    for (int nt = 0; nt < 2; ++nt)
#pragma unroll
      for (int i = 0; i < 8; ++i)
        lds[(half8 + i) * 40 + nt * 16 + col16] = (_Float16)S[nt][i];
    __syncthreads();
    const v16h pa =
        load2x8(&lds[col16 * 40 + half8], &lds[col16 * 40 + 16 + half8]);
    __syncthreads();

    // ---- O += P @ V (V is transposed: contiguous in t) ----------------
#pragma unroll
    for (int a = 0; a < 4; ++a) {
      const _Float16* vrow = vbase + (size_t)(a * 16 + col16) * kT + ks;
      O[a] = wmma16(pa, load2x8(vrow + half8, vrow + 16 + half8), O[a]);
    }
  }

  // normalize and store as (b, t, h*64+d) f16 row-major (GEMM-A ready)
#pragma unroll
  for (int a = 0; a < 4; ++a) {
    const int d = a * 16 + col16;
#pragma unroll
    for (int i = 0; i < 8; ++i) {
      const int t = qb + half8 + i;
      attn[((size_t)(b * kT + t)) * kC + h * kD + d] =
          (_Float16)(O[a][i] / lrun[i]);
    }
  }
}

// ---------------------------------------------------------------------------
// Kernel 3: out = attn @ W_proj + b_proj (fp32 out). One wave -> 32x64 tile.
// grid = (256, 3), block 128.
// ---------------------------------------------------------------------------
__global__ __launch_bounds__(128) void proj_kernel(
    const _Float16* __restrict__ attn, const _Float16* __restrict__ WpT,
    const float* __restrict__ bias, float* __restrict__ out) {
  const int lane  = threadIdx.x & 31;
  const int wave  = threadIdx.x >> 5;
  const int col16 = lane & 15;
  const int half8 = (lane >> 4) * 8;
  const int hh16  = (lane >> 4) * 16;
  const int g0    = blockIdx.x * 32;
  const int nb    = (blockIdx.y * 4 + wave) * kD;

  v8f acc[2][4] = {};
  const _Float16* arow0 = attn + (size_t)(g0 + col16) * kC;
  const _Float16* arow1 = attn + (size_t)(g0 + 16 + col16) * kC;
  for (int kb = 0; kb < kC; kb += 32) {
    const v16h a0 = load2x8(arow0 + kb + half8, arow0 + kb + 16 + half8);
    const v16h a1 = load2x8(arow1 + kb + half8, arow1 + kb + 16 + half8);
#pragma unroll
    for (int nt = 0; nt < 4; ++nt) {
      const _Float16* wp =
          WpT + (size_t)(nb + nt * 16 + col16) * kC + kb + half8;
      const v16h bf = load2x8(wp, wp + 16);
      acc[0][nt] = wmma16(a0, bf, acc[0][nt]);
      acc[1][nt] = wmma16(a1, bf, acc[1][nt]);
    }
  }
#pragma unroll
  for (int mt = 0; mt < 2; ++mt)
#pragma unroll
    for (int nt = 0; nt < 4; ++nt) {
      const int n = nb + nt * 16 + col16;
      const float bv = bias[n];
#pragma unroll
      for (int i = 0; i < 8; ++i)
        out[(size_t)(g0 + mt * 16 + half8 + i) * kC + n] = acc[mt][nt][i] + bv;
    }
}

// ---------------------------------------------------------------------------
extern "C" void kernel_launch(void* const* d_in, const int* in_sizes, int n_in,
                              void* d_out, int out_size, void* d_ws,
                              size_t ws_size, hipStream_t stream) {
  (void)in_sizes; (void)n_in; (void)out_size; (void)ws_size;
  const float* x     = (const float*)d_in[0];
  const float* fcos  = (const float*)d_in[1];
  const float* fsin  = (const float*)d_in[2];
  const float* Watt  = (const float*)d_in[3];
  const float* Wproj = (const float*)d_in[4];
  const float* bproj = (const float*)d_in[5];

  _Float16* xh     = (_Float16*)d_ws;                    // 6,291,456
  _Float16* WattT  = xh + kXElems;                       // 1,769,472
  _Float16* WprojT = WattT + (size_t)kQKV * kC;          //   589,824
  _Float16* qws    = WprojT + (size_t)kC * kC;
  _Float16* kws    = qws + kHeadElems;
  _Float16* vtws   = kws + kHeadElems;
  _Float16* attnws = vtws + kHeadElems;

  // one-time precision/layout prep
  cvt_f16_kernel<<<(int)(kXElems / 8 / 256), 256, 0, stream>>>(
      x, xh, (int)(kXElems / 8));
  transpose_cvt_kernel<<<(kC * kQKV + 255) / 256, 256, 0, stream>>>(
      Watt, WattT, kC, kQKV);
  transpose_cvt_kernel<<<(kC * kC + 255) / 256, 256, 0, stream>>>(
      Wproj, WprojT, kC, kC);

  qkv_rope_kernel<<<dim3(256, 9), 128, 0, stream>>>(xh, fcos, fsin, WattT, qws,
                                                    kws, vtws);
  attn_kernel<<<dim3(128, 48), 32, 0, stream>>>(qws, kws, vtws, attnws);
  proj_kernel<<<dim3(256, 3), 128, 0, stream>>>(attnws, WprojT, bproj,
                                                (float*)d_out);
}
